// CL_81750407512203
// MI455X (gfx1250) — matrix operand order repaired and checked by reading, not compiled, not run on hardware
//
#include <hip/hip_runtime.h>

typedef __attribute__((ext_vector_type(16))) _Float16 v16h;
typedef __attribute__((ext_vector_type(8)))  float    v8f;

// Problem constants
#define BB    8
#define CC    128
#define HH    192
#define WW    192
#define PATCH 11
#define NP    121         // PATCH*PATCH

// ---------------------------------------------------------------------------
// Kernel 1: correlation via banded WMMA GEMM with async-to-LDS double buffer.
// One block per (b, output row i); the x0 row is staged once; each of the 11
// x1 rows is DMA'd into LDS with GLOBAL_LOAD_ASYNC_TO_LDS_B128 one di ahead
// of the WMMA consuming it. 211 KB static LDS (CDNA5 WGP has 320 KB).
// corr[b, diIdx*11 + pj, i, j] = sum_c x0[b,c,i,j] * x1[b,c,i+di, j+dj]
// ---------------------------------------------------------------------------
__global__ __launch_bounds__(256) void corr_wmma_kernel(
    const float* __restrict__ x0, const float* __restrict__ x1,
    _Float16* __restrict__ corr)
{
    constexpr int STRH = 136;             // LDS col stride in halves (272B, 16B-aligned)
    __shared__ _Float16 x0s[192 * STRH];  // [col][c] f16            52,224 B
    __shared__ _Float16 x1s[224 * STRH];  // [16+col][c] f16 (pads)  60,928 B
    __shared__ float    stg[CC * WW];     // async fp32 stage [c][w] 98,304 B

    const int tid   = threadIdx.x;
    const int lane  = tid & 31;
    const int wv    = tid >> 5;     // wave id 0..7
    const int nLane = lane & 15;    // tile-local column index
    const int hi    = lane >> 4;    // lane half

    const int irow = blockIdx.x % HH;
    const int b    = blockIdx.x / HH;
    const float* x1base = x1 + (size_t)b * CC * HH * WW;

    // Zero the spatial-pad columns of x1s (cols 0..15 and 208..223), all c.
    for (int e = tid; e < 32 * STRH; e += 256) {
        int pc = e / STRH;
        int cc = (pc < 16) ? pc : (192 + pc);
        x1s[cc * STRH + (e - pc * STRH)] = (_Float16)0.0f;
    }

    // Stage x0 row once: fp32 -> f16, transpose to [col][c]. float4 loads.
    {
        const float* src = x0 + ((size_t)b * CC * HH + irow) * WW;
        for (int e = tid; e < CC * 48; e += 256) {
            int c = e / 48, w4 = e - c * 48;
            const float4 f = *(const float4*)(src + (size_t)c * HH * WW + w4 * 4);
            _Float16* d = &x0s[(w4 * 4) * STRH + c];
            d[0 * STRH] = (_Float16)f.x;
            d[1 * STRH] = (_Float16)f.y;
            d[2 * STRH] = (_Float16)f.z;
            d[3 * STRH] = (_Float16)f.w;
        }
    }

    // Issue async DMA of the first x1 row (di = -10) into stg.
    {
        const int i1 = irow - 10;
        if (i1 >= 0 && i1 < HH) {
            const float* src = x1base + (size_t)i1 * WW;
            for (int e = tid; e < CC * 48; e += 256) {
                int c = e / 48, w4 = e - c * 48;
                unsigned long long ga =
                    (unsigned long long)(src + (size_t)c * HH * WW + w4 * 4);
                unsigned int la = (unsigned int)(size_t)&stg[c * WW + w4 * 4];
                asm volatile("global_load_async_to_lds_b128 %0, %1, off"
                             :: "v"(la), "v"(ga) : "memory");
            }
        }
    }
    __syncthreads();

    for (int diIdx = 0; diIdx < PATCH; ++diIdx) {
        const int  i1  = irow + (diIdx - 5) * 2;
        const bool inR = (i1 >= 0) && (i1 < HH);

        // Drain this wave's async ops; barrier makes all waves' DMA visible.
        asm volatile("s_wait_asynccnt 0x0" ::: "memory");
        __syncthreads();

        // Convert staged fp32 row -> f16 transposed WMMA layout (zero if OOB).
        for (int e = tid; e < CC * 48; e += 256) {
            int c = e / 48, w4 = e - c * 48;
            float4 f = {0.f, 0.f, 0.f, 0.f};
            if (inR) f = *(const float4*)&stg[c * WW + w4 * 4];
            _Float16* d = &x1s[(16 + w4 * 4) * STRH + c];
            d[0 * STRH] = (_Float16)f.x;
            d[1 * STRH] = (_Float16)f.y;
            d[2 * STRH] = (_Float16)f.z;
            d[3 * STRH] = (_Float16)f.w;
        }
        __syncthreads();   // convert done: stg reusable, x1s readable

        // Issue async DMA of the next x1 row while the WMMAs below run.
        {
            const int i1n = irow + (diIdx - 4) * 2;
            if (diIdx + 1 < PATCH && i1n >= 0 && i1n < HH) {
                const float* src = x1base + (size_t)i1n * WW;
                for (int e = tid; e < CC * 48; e += 256) {
                    int c = e / 48, w4 = e - c * 48;
                    unsigned long long ga =
                        (unsigned long long)(src + (size_t)c * HH * WW + w4 * 4);
                    unsigned int la = (unsigned int)(size_t)&stg[c * WW + w4 * 4];
                    asm volatile("global_load_async_to_lds_b128 %0, %1, off"
                                 :: "v"(la), "v"(ga) : "memory");
                }
            }
            // L2 prefetch two rows ahead so the next async issue hits cache.
            const int i2n = irow + (diIdx - 3) * 2;
            if (diIdx + 2 < PATCH && i2n >= 0 && i2n < HH) {
                const char* nsrc = (const char*)(x1base + (size_t)i2n * WW);
                for (int e = tid; e < 768; e += 256)   // 128 ch x 6 cachelines
                    __builtin_prefetch(
                        nsrc + (size_t)(e / 6) * (HH * WW * 4) + (e % 6) * 128, 0, 1);
            }
        }

        // 12 j-tiles x 3 band tiles = 36 tile products over 8 waves; K=128.
        #pragma unroll
        for (int li = 0; li < 5; ++li) {
            int tt = wv + li * 8;
            if (tt < 36) {
                int jt   = tt / 3;
                int tsel = tt - jt * 3;                       // band shift t = (tsel-1)*16
                const _Float16* pa = &x1s[(jt * 16 + tsel * 16 + nLane) * STRH];
                const _Float16* pb = &x0s[(jt * 16 + nLane) * STRH];
                v8f acc = {0.f, 0.f, 0.f, 0.f, 0.f, 0.f, 0.f, 0.f};
                #pragma unroll
                for (int kk = 0; kk < 4; ++kk) {              // 4 x K=32 chunks
                    v16h a, bf;
                    int kb = kk * 32 + hi * 8;                // A: {kb..kb+7} U {kb+16..kb+23}
                    #pragma unroll
                    for (int q = 0; q < 8; ++q) {
                        a[q]     = pa[kb + q];
                        a[8 + q] = pa[kb + 16 + q];
                    }
                    int kc = kk * 32 + hi * 8;                // B: {kc..kc+7} U {kc+16..kc+23}
                    #pragma unroll
                    for (int q = 0; q < 8; ++q) { bf[q] = pb[kc + q]; bf[8 + q] = pb[kc + 16 + q]; }
                    acc = __builtin_amdgcn_wmma_f32_16x16x32_f16(
                        false, a, false, bf, (short)0, acc, false, false);
                }
                // Extract the even diagonals (dj = m - n + t, |dj| <= 10), store f16.
                int col = jt * 16 + nLane;
                #pragma unroll
                for (int v = 0; v < 8; ++v) {
                    int m  = v + hi * 8;
                    int dj = m - nLane + (tsel - 1) * 16;
                    if (((dj & 1) == 0) && dj >= -10 && dj <= 10) {
                        int p = diIdx * PATCH + (dj / 2 + 5);
                        corr[(((size_t)b * NP + p) * HH + irow) * WW + col] =
                            (_Float16)acc[v];
                    }
                }
            }
        }
        // Next iteration's s_wait_asynccnt + barrier orders compute vs. restage.
    }
}

// ---------------------------------------------------------------------------
// Kernel 2: 3x3 conv 121->8 + bias, accumulate per-channel sum / sumsq for BN.
// One block per (b, 16x16 output tile). corr read as f16.
// ---------------------------------------------------------------------------
__global__ __launch_bounds__(256) void conv_stats_kernel(
    const _Float16* __restrict__ corr, const float* __restrict__ wgt,
    const float* __restrict__ bias, float* __restrict__ convout,
    float* __restrict__ stats)
{
    __shared__ float wl[8 * NP * 9];   // 34,848 B
    __shared__ float bl[8];
    __shared__ float tile[18 * 18];
    __shared__ float bstat[16];

    const int tid = threadIdx.x;
    const int b   = blockIdx.z;
    const int ty  = blockIdx.y, tx = blockIdx.x;

    for (int e = tid; e < 8 * NP * 9; e += 256) wl[e] = wgt[e];
    if (tid < 8)  bl[tid]    = bias[tid];
    if (tid < 16) bstat[tid] = 0.0f;
    __syncthreads();

    const int ti = tid >> 4, tj = tid & 15;
    float acc[8] = {0.f, 0.f, 0.f, 0.f, 0.f, 0.f, 0.f, 0.f};

    for (int p = 0; p < NP; ++p) {
        __syncthreads();
        for (int e = tid; e < 324; e += 256) {
            int ei = e / 18, ej = e - ei * 18;
            int gi = ty * 16 - 1 + ei, gj = tx * 16 - 1 + ej;
            float v = 0.0f;
            if (gi >= 0 && gi < HH && gj >= 0 && gj < WW)
                v = (float)corr[(((size_t)b * NP + p) * HH + gi) * WW + gj];
            tile[e] = v;
        }
        __syncthreads();
        #pragma unroll
        for (int ki = 0; ki < 3; ++ki)
            #pragma unroll
            for (int kj = 0; kj < 3; ++kj) {
                float v = tile[(ti + ki) * 18 + (tj + kj)];
                const float* wp = &wl[p * 9 + ki * 3 + kj];
                #pragma unroll
                for (int o = 0; o < 8; ++o)
                    acc[o] = fmaf(wp[o * NP * 9], v, acc[o]);
            }
    }

    int gi = ty * 16 + ti, gj = tx * 16 + tj;
    #pragma unroll
    for (int o = 0; o < 8; ++o) {
        float v = acc[o] + bl[o];
        convout[(((size_t)b * 8 + o) * HH + gi) * WW + gj] = v;
        atomicAdd(&bstat[o], v);
        atomicAdd(&bstat[8 + o], v * v);
    }
    __syncthreads();
    if (tid < 16) atomicAdd(&stats[tid], bstat[tid]);
}

// ---------------------------------------------------------------------------
// Kernel 3: BatchNorm (batch stats, biased var) + affine + ReLU.
// ---------------------------------------------------------------------------
__global__ __launch_bounds__(256) void bn_relu_kernel(
    const float* __restrict__ convout, const float* __restrict__ stats,
    const float* __restrict__ gamma, const float* __restrict__ beta,
    float* __restrict__ out, int n)
{
    int idx = blockIdx.x * 256 + threadIdx.x;
    if (idx >= n) return;
    const int HW = HH * WW;
    int o = (idx / HW) & 7;
    const float cnt = (float)(BB * HW);      // 294912
    float mean = stats[o] / cnt;
    float var  = stats[8 + o] / cnt - mean * mean;
    float sc   = gamma[o] * rsqrtf(var + 1e-5f);
    float v    = (convout[idx] - mean) * sc + beta[o];
    out[idx]   = fmaxf(v, 0.0f);
}

// ---------------------------------------------------------------------------
extern "C" void kernel_launch(void* const* d_in, const int* in_sizes, int n_in,
                              void* d_out, int out_size, void* d_ws, size_t ws_size,
                              hipStream_t stream)
{
    const float* x0     = (const float*)d_in[0];
    const float* x1     = (const float*)d_in[1];
    const float* conv_w = (const float*)d_in[2];
    const float* conv_b = (const float*)d_in[3];
    const float* gamma  = (const float*)d_in[4];
    const float* beta   = (const float*)d_in[5];
    float* out = (float*)d_out;

    const size_t corrElems = (size_t)BB * NP * HH * WW;  // 35,684,352 (f16 -> 71.4 MB)
    const size_t convElems = (size_t)BB * 8 * HH * WW;   //  2,359,296

    _Float16* corr  = (_Float16*)d_ws;
    float* convout  = (float*)((char*)d_ws + corrElems * sizeof(_Float16));
    float* stats    = convout + convElems;

    hipMemsetAsync(stats, 0, 16 * sizeof(float), stream);

    corr_wmma_kernel<<<dim3(BB * HH), dim3(256), 0, stream>>>(x0, x1, corr);
    conv_stats_kernel<<<dim3(WW / 16, HH / 16, BB), dim3(256), 0, stream>>>(
        corr, conv_w, conv_b, convout, stats);
    int n = (int)convElems;
    bn_relu_kernel<<<dim3((n + 255) / 256), dim3(256), 0, stream>>>(
        convout, stats, gamma, beta, out, n);
}
